// MyRNN_77653008712316
// MI455X (gfx1250) — matrix operand order repaired and checked by reading, not compiled, not run on hardware
//
#include <hip/hip_runtime.h>
#include <hip/hip_bf16.h>

// ---------------------------------------------------------------------------
// RNN: wx = x @ W_w^T + W_b  (one big bf16 WMMA GEMM, M=32768,N=K=1024)
// then 512 sequential steps: h = tanh(wx_t + h @ U_w^T + U_b + bias)
// Each step is a lean LDS-free WMMA GEMM; h is ping-ponged in bf16 workspace.
// ---------------------------------------------------------------------------

typedef __attribute__((ext_vector_type(16))) __bf16 v16bf;
typedef __attribute__((ext_vector_type(8)))  float  v8f;

union BF16x16 {
    v16bf v;
    uint4 q[2];
};

__device__ __forceinline__ v8f wmma_bf16(v16bf a, v16bf b, v8f c) {
    // (neg_a, A, neg_b, B, c_mod, C, reuse_a, reuse_b)
    return __builtin_amdgcn_wmma_f32_16x16x32_bf16(false, a, false, b,
                                                   (short)0, c, false, false);
}

// A fragment: 16x32 bf16, source row-major MxK (bf16 in global).
// ISA layout: lane<16 holds M=lane, K={0..7, 16..23}; lane>=16 -> K base 8.
__device__ __forceinline__ v16bf load_a_bf16(const __bf16* __restrict__ A,
                                             int ld, int m0, int kk, int lane) {
    int m  = m0 + (lane & 15);
    int kb = (lane >> 4) * 8;
    const __bf16* p = A + (size_t)m * ld + kk + kb;
    BF16x16 r;
    r.q[0] = *(const uint4*)(p);        // K = kb .. kb+7
    r.q[1] = *(const uint4*)(p + 16);   // K = kb+16 .. kb+23
    return r.v;
}

// A fragment from f32 source: load 16 floats, convert to bf16 in registers.
__device__ __forceinline__ v16bf load_a_f32(const float* __restrict__ A,
                                            int ld, int m0, int kk, int lane) {
    int m  = m0 + (lane & 15);
    int kb = (lane >> 4) * 8;
    const float* p = A + (size_t)m * ld + kk + kb;
    float4 f0 = *(const float4*)(p);
    float4 f1 = *(const float4*)(p + 4);
    float4 f2 = *(const float4*)(p + 16);
    float4 f3 = *(const float4*)(p + 20);
    BF16x16 r;
    __bf16* e = (__bf16*)&r;
    e[0]  = (__bf16)f0.x; e[1]  = (__bf16)f0.y; e[2]  = (__bf16)f0.z; e[3]  = (__bf16)f0.w;
    e[4]  = (__bf16)f1.x; e[5]  = (__bf16)f1.y; e[6]  = (__bf16)f1.z; e[7]  = (__bf16)f1.w;
    e[8]  = (__bf16)f2.x; e[9]  = (__bf16)f2.y; e[10] = (__bf16)f2.z; e[11] = (__bf16)f2.w;
    e[12] = (__bf16)f3.x; e[13] = (__bf16)f3.y; e[14] = (__bf16)f3.z; e[15] = (__bf16)f3.w;
    return r.v;
}

// B fragment: 32x16 bf16 where B[k][n] = W[n][k] (W row-major NxK) -> B column
// n is W row n, contiguous in K. lane<16: n=lane, K={0..7, 16..23}; lane>=16: K base 8.
__device__ __forceinline__ v16bf load_b_bf16(const __bf16* __restrict__ W,
                                             int ld, int n0, int kk, int lane) {
    int n  = n0 + (lane & 15);
    int kb = (lane >> 4) * 8;
    const __bf16* p = W + (size_t)n * ld + kk + kb;
    BF16x16 r;
    r.q[0] = *(const uint4*)(p);       // K = kb .. kb+7
    r.q[1] = *(const uint4*)(p + 16);  // K = kb+16 .. kb+23
    return r.v;
}

// ---------------------------------------------------------------------------
// Kernel 0: convert W_w and U_w (f32 DxD) to bf16 in workspace.
// ---------------------------------------------------------------------------
__global__ __launch_bounds__(256) void cvt_weights(const float* __restrict__ w,
                                                   const float* __restrict__ u,
                                                   __bf16* __restrict__ wo,
                                                   __bf16* __restrict__ uo,
                                                   int n) {
    int base = (blockIdx.x * blockDim.x + threadIdx.x) * 4;
    if (base < n) {
        float4 fw = *(const float4*)(w + base);
        float4 fu = *(const float4*)(u + base);
        wo[base + 0] = (__bf16)fw.x; wo[base + 1] = (__bf16)fw.y;
        wo[base + 2] = (__bf16)fw.z; wo[base + 3] = (__bf16)fw.w;
        uo[base + 0] = (__bf16)fu.x; uo[base + 1] = (__bf16)fu.y;
        uo[base + 2] = (__bf16)fu.z; uo[base + 3] = (__bf16)fu.w;
    }
}

// ---------------------------------------------------------------------------
// Kernel 1: wx = x @ W_w^T + W_b, written into d_out's resultEmbs region.
// 8 waves/block; block tile 32(M) x 256(N); each wave computes 16x64
// (4 WMMA tiles sharing one A fragment -> 4 cvts per WMMA instead of 8).
// grid = (N/256, M/32).
// ---------------------------------------------------------------------------
__global__ __launch_bounds__(256) void wx_gemm(const float* __restrict__ x,
                                               const __bf16* __restrict__ Wbf,
                                               const float* __restrict__ W_b,
                                               float* __restrict__ wx) {
    const int D = 1024;
    int lane = threadIdx.x & 31;
    int wave = threadIdx.x >> 5;              // wave32
    int m0 = blockIdx.y * 32 + (wave & 1) * 16;
    int n0 = blockIdx.x * 256 + (wave >> 1) * 64;

    v8f acc0 = {};
    v8f acc1 = {};
    v8f acc2 = {};
    v8f acc3 = {};
    for (int kk = 0; kk < D; kk += 32) {
        v16bf a  = load_a_f32(x, D, m0, kk, lane);
        v16bf b0 = load_b_bf16(Wbf, D, n0,      kk, lane);
        v16bf b1 = load_b_bf16(Wbf, D, n0 + 16, kk, lane);
        v16bf b2 = load_b_bf16(Wbf, D, n0 + 32, kk, lane);
        v16bf b3 = load_b_bf16(Wbf, D, n0 + 48, kk, lane);
        acc0 = wmma_bf16(a, b0, acc0);
        acc1 = wmma_bf16(a, b1, acc1);
        acc2 = wmma_bf16(a, b2, acc2);
        acc3 = wmma_bf16(a, b3, acc3);
    }

    int col = lane & 15, grp = lane >> 4;
    int gn0 = n0 + col, gn1 = gn0 + 16, gn2 = gn0 + 32, gn3 = gn0 + 48;
    float c0 = W_b[gn0], c1 = W_b[gn1], c2 = W_b[gn2], c3 = W_b[gn3];
#pragma unroll
    for (int v = 0; v < 8; ++v) {
        int m = m0 + v + 8 * grp;             // C/D layout: M = v + 8*group
        size_t row = (size_t)m * D;
        wx[row + gn0] = acc0[v] + c0;
        wx[row + gn1] = acc1[v] + c1;
        wx[row + gn2] = acc2[v] + c2;
        wx[row + gn3] = acc3[v] + c3;
    }
}

// ---------------------------------------------------------------------------
// Kernel 2 (x512): one recurrence step.
//   h[b,j] = tanh(wx[b,t,j] + sum_k hprev[b,k]*U[j,k] + U_b[j] + bias[j])
// seq holds wx for slice t on entry; overwritten in place with h.
// hprev/hnext are bf16 ping-pong buffers (WMMA-ready, no LDS needed).
// 4 waves/block, block tile 16(M) x 64(N); grid = (N/64, B/16).
// One 16x16 tile per wave: maximizes wave count for the latency-bound scan.
// ---------------------------------------------------------------------------
__global__ __launch_bounds__(128) void rnn_step(float* __restrict__ seq,
                                                const __bf16* __restrict__ Ubf,
                                                const float* __restrict__ U_b,
                                                const float* __restrict__ bias,
                                                const __bf16* __restrict__ hprev,
                                                __bf16* __restrict__ hnext,
                                                float* __restrict__ hT,
                                                int t, int hasPrev) {
    const int D = 1024, T = 512;
    int lane = threadIdx.x & 31;
    int wave = threadIdx.x >> 5;
    int m0 = blockIdx.y * 16;                 // batch rows
    int n0 = blockIdx.x * 64 + wave * 16;     // hidden cols

    v8f acc = {};
    if (hasPrev) {
        for (int kk = 0; kk < D; kk += 32) {
            // prefetch next chunk of the U rows this wave consumes
            if (kk + 32 < D)
                __builtin_prefetch(Ubf + (size_t)(n0 + (lane & 15)) * D + kk + 32, 0, 3);
            v16bf a = load_a_bf16(hprev, D, m0, kk, lane);
            v16bf b = load_b_bf16(Ubf,   D, n0, kk, lane);
            acc = wmma_bf16(a, b, acc);
        }
    }

    int col = lane & 15, grp = lane >> 4;
    int gn = n0 + col;
    float cadd = U_b[gn] + bias[gn];
#pragma unroll
    for (int v = 0; v < 8; ++v) {
        int b = m0 + v + 8 * grp;             // batch index
        size_t idx = ((size_t)b * T + t) * D + gn;
        float h = tanhf(acc[v] + seq[idx] + cadd);
        seq[idx] = h;                         // resultEmbs[b,t,:]
        hnext[(size_t)b * D + gn] = (__bf16)h;
        if (hT) hT[(size_t)b * D + gn] = h;   // final hidden state tail
    }
}

// ---------------------------------------------------------------------------
extern "C" void kernel_launch(void* const* d_in, const int* in_sizes, int n_in,
                              void* d_out, int out_size, void* d_ws, size_t ws_size,
                              hipStream_t stream) {
    const int B = 64, T = 512, D = 1024;
    const float* x    = (const float*)d_in[0];
    const float* W_w  = (const float*)d_in[1];
    const float* W_b  = (const float*)d_in[2];
    const float* U_w  = (const float*)d_in[3];
    const float* U_b  = (const float*)d_in[4];
    const float* bias = (const float*)d_in[5];
    float* out = (float*)d_out;

    // workspace layout (bf16): W (D*D) | U (D*D) | h ping (B*D) | h pong (B*D)
    __bf16* Wbf = (__bf16*)d_ws;
    __bf16* Ubf = Wbf + (size_t)D * D;
    __bf16* h0  = Ubf + (size_t)D * D;
    __bf16* h1  = h0 + (size_t)B * D;

    // 0) weights f32 -> bf16 (1M elems each, 4 per thread)
    cvt_weights<<<(D * D) / (256 * 4), 256, 0, stream>>>(W_w, U_w, Wbf, Ubf, D * D);

    // 1) wx = x @ W_w^T + W_b directly into resultEmbs region of d_out
    wx_gemm<<<dim3(D / 256, (B * T) / 32), 256, 0, stream>>>(x, Wbf, W_b, out);

    // 2) sequential scan: 512 dependent step kernels
    float* tail = out + (size_t)B * T * D;    // hT region (B,1,D)
    for (int t = 0; t < T; ++t) {
        const __bf16* hp = (t & 1) ? h1 : h0;
        __bf16*       hn = (t & 1) ? h0 : h1;
        rnn_step<<<dim3(D / 64, B / 16), 128, 0, stream>>>(
            out, Ubf, U_b, bias, hp, hn, (t == T - 1) ? tail : nullptr, t, t > 0);
    }
}
